// RoPETransformerEncoderLayer_412316860449
// MI455X (gfx1250) — matrix operand order repaired and checked by reading, not compiled, not run on hardware
//
#include <hip/hip_runtime.h>

typedef _Float16 v16h __attribute__((ext_vector_type(16)));
typedef _Float16 v8h  __attribute__((ext_vector_type(8)));
typedef float    v8f  __attribute__((ext_vector_type(8)));
typedef int      v4i  __attribute__((ext_vector_type(4)));

__device__ __forceinline__ v8f wmma_f16(v16h a, v16h b, v8f c) {
    return __builtin_amdgcn_wmma_f32_16x16x32_f16(
        false, a, false, b, (short)0, c, false, false);
}

// ---------------- CDNA5 async global->LDS copy (ASYNCcnt path), with fallback
#if defined(__HIP_DEVICE_COMPILE__) && __has_builtin(__builtin_amdgcn_global_load_async_to_lds_b128)
#define HAVE_ASYNC_LDS 1
#else
#define HAVE_ASYNC_LDS 0
#endif

__device__ __forceinline__ void async_copy16(const _Float16* g, _Float16* l) {
#if HAVE_ASYNC_LDS
    __builtin_amdgcn_global_load_async_to_lds_b128(
        (__attribute__((address_space(1))) v4i*)g,
        (__attribute__((address_space(3))) v4i*)l, 0, 0);
#else
    *reinterpret_cast<uint4*>(l) = *reinterpret_cast<const uint4*>(g);
#endif
}

template <int N>
__device__ __forceinline__ void wait_asynccnt() {
#if defined(__HIP_DEVICE_COMPILE__)
#if __has_builtin(__builtin_amdgcn_s_wait_asynccnt)
    __builtin_amdgcn_s_wait_asynccnt(N);
#else
    asm volatile("s_wait_asynccnt %0" ::"i"(N) : "memory");
#endif
#endif
}

// ---------------------------------------------------------------- transpose+cast
// src f32 [K][N] -> dst f16 [N][K]
__global__ __launch_bounds__(256)
void transpose_cast(const float* __restrict__ src, _Float16* __restrict__ dst,
                    int K, int N) {
    __shared__ float tile[32][33];
    int kb = blockIdx.y * 32, nb = blockIdx.x * 32;
    int tx = threadIdx.x & 31, ty = threadIdx.x >> 5;   // 32 x 8
#pragma unroll
    for (int i = 0; i < 32; i += 8)
        tile[ty + i][tx] = src[(size_t)(kb + ty + i) * N + nb + tx];
    __syncthreads();
#pragma unroll
    for (int i = 0; i < 32; i += 8)
        dst[(size_t)(nb + ty + i) * K + kb + tx] = (_Float16)tile[tx][ty + i];
}

// ---------------------------------------------------------------- LayerNorm -> f16
__global__ __launch_bounds__(256)
void ln_cast(const float* __restrict__ X, const float* __restrict__ gamma,
             const float* __restrict__ beta, _Float16* __restrict__ H, int D) {
    int row = blockIdx.x;
    const float* xr = X + (size_t)row * D;
    __shared__ float red0[256];
    __shared__ float red1[256];
    float s = 0.f, s2 = 0.f;
    for (int i = threadIdx.x; i < D; i += 256) {
        float v = xr[i]; s += v; s2 += v * v;
    }
    red0[threadIdx.x] = s; red1[threadIdx.x] = s2;
    __syncthreads();
    for (int off = 128; off > 0; off >>= 1) {
        if ((int)threadIdx.x < off) {
            red0[threadIdx.x] += red0[threadIdx.x + off];
            red1[threadIdx.x] += red1[threadIdx.x + off];
        }
        __syncthreads();
    }
    float mu  = red0[0] / (float)D;
    float var = red1[0] / (float)D - mu * mu;
    float rs  = rsqrtf(var + 1e-5f);
    for (int i = threadIdx.x; i < D; i += 256) {
        float v = (xr[i] - mu) * rs * gamma[i] + beta[i];
        H[(size_t)row * D + i] = (_Float16)v;
    }
}

// ---------------------------------------------------------------- tiled WMMA GEMM
// C[M,N] = A[M,K](f16,row) x Bt[N,K](f16, pre-transposed)   + epilogue
// mode 0: Cf = acc ; 1: Cf = acc+resid ; 2: Ch = gelu(acc+bias) ; 3: Cf = acc+bias+resid
#define BM 128
#define BN 128
#define BK 32

__global__ __launch_bounds__(256)
void gemm_f16(const _Float16* __restrict__ A, const _Float16* __restrict__ Bt,
              const float* __restrict__ bias, const float* __restrict__ resid,
              float* __restrict__ Cf, _Float16* __restrict__ Ch,
              int M, int N, int K, int mode) {
    __shared__ __align__(32) _Float16 As[2][BM * BK];
    __shared__ __align__(32) _Float16 Bs[2][BN * BK];

    const int tid  = threadIdx.x;
    const int lane = tid & 31;
    const int w    = tid >> 5;
    const int g    = lane >> 4;
    const int ln   = lane & 15;
    const int m0   = blockIdx.y * BM;
    const int n0   = blockIdx.x * BN;
    const int wm0  = (w >> 1) * 32;     // 4 waves down (32 rows each)
    const int wn0  = (w & 1) * 64;      // 2 waves across (64 cols each)

    v8f acc[2][4] = {};

    // each thread moves 2x16B of A and 2x16B of B per tile (128x32 f16 each)
    auto issue_tile = [&](int k0, int buf) {
#pragma unroll
        for (int i = 0; i < 2; ++i) {
            int id = tid + 256 * i;
            int r = id >> 2, seg = (id & 3) * 8;
            async_copy16(A + (size_t)(m0 + r) * K + k0 + seg,
                         &As[buf][r * BK + seg]);
        }
#pragma unroll
        for (int i = 0; i < 2; ++i) {
            int id = tid + 256 * i;
            int r = id >> 2, seg = (id & 3) * 8;
            async_copy16(Bt + (size_t)(n0 + r) * K + k0 + seg,
                         &Bs[buf][r * BK + seg]);
        }
    };

    const int nk = K / BK;
    issue_tile(0, 0);
    int buf = 0;
    for (int kt = 0; kt < nk; ++kt) {
        if (kt + 1 < nk) {
            issue_tile((kt + 1) * BK, buf ^ 1);
            wait_asynccnt<4>();          // tile kt finished, next still in flight
        } else {
            wait_asynccnt<0>();
        }
        __syncthreads();

        v16h af[2], bf[4];
#pragma unroll
        for (int m = 0; m < 2; ++m) {
            int row = wm0 + 16 * m + ln;
            const _Float16* base = &As[buf][row * BK];
            v8h lo = *reinterpret_cast<const v8h*>(base + 8 * g);
            v8h hi = *reinterpret_cast<const v8h*>(base + 16 + 8 * g);
#pragma unroll
            for (int j = 0; j < 8; ++j) { af[m][j] = lo[j]; af[m][j + 8] = hi[j]; }
        }
#pragma unroll
        for (int n = 0; n < 4; ++n) {
            int col = wn0 + 16 * n + ln;
            v8h blo = *reinterpret_cast<const v8h*>(&Bs[buf][col * BK + 8 * g]);
            v8h bhi = *reinterpret_cast<const v8h*>(&Bs[buf][col * BK + 16 + 8 * g]);
#pragma unroll
            for (int j = 0; j < 8; ++j) { bf[n][j] = blo[j]; bf[n][j + 8] = bhi[j]; }
        }
#pragma unroll
        for (int m = 0; m < 2; ++m)
#pragma unroll
            for (int n = 0; n < 4; ++n)
                acc[m][n] = wmma_f16(af[m], bf[n], acc[m][n]);

        __syncthreads();                 // all waves done with buf before refill
        buf ^= 1;
    }

#pragma unroll
    for (int m = 0; m < 2; ++m) {
#pragma unroll
        for (int n = 0; n < 4; ++n) {
            int col = n0 + wn0 + 16 * n + ln;
            float bv = (mode >= 2) ? bias[col] : 0.f;
#pragma unroll
            for (int r = 0; r < 8; ++r) {
                int row = m0 + wm0 + 16 * m + r + 8 * g;
                size_t idx = (size_t)row * N + col;
                float v = acc[m][n][r];
                if (mode == 0) {
                    Cf[idx] = v;
                } else if (mode == 1) {
                    Cf[idx] = v + resid[idx];
                } else if (mode == 2) {
                    float t  = v + bv;
                    float gl = 0.5f * t * (1.f + erff(t * 0.70710678118654752f));
                    Ch[idx] = (_Float16)gl;
                } else {
                    Cf[idx] = v + bv + resid[idx];
                }
            }
        }
    }
}

// ---------------------------------------------------------------- RoPE + relayout
// qkvf f32 [B,S,3*D] (q|k|v). Outputs: q16,k16 f16 [B,H,S,HD]; vT16 f16 [B,H,HD,S]
__global__ void rope_cast(const float* __restrict__ qkvf,
                          _Float16* __restrict__ q16, _Float16* __restrict__ k16,
                          _Float16* __restrict__ vT16, int B, int H, int S) {
    const int HD = 64, D = 1024;
    int idx = blockIdx.x * blockDim.x + threadIdx.x;   // over B*S*H
    int h = idx % H;
    int s = (idx / H) % S;
    int b = idx / (H * S);
    if (b >= B) return;
    const float* base = qkvf + (size_t)(b * S + s) * (3 * D) + h * HD;
    const float* qr = base;
    const float* kr = base + D;
    const float* vr = base + 2 * D;
    const size_t o16 = ((size_t)(b * H + h) * S + s) * HD;
    const float ln_base = 9.210340371976184f;   // ln(10000)
#pragma unroll 4
    for (int d = 0; d < 32; ++d) {
        float inv = expf(-(2.f * d / (float)HD) * ln_base);
        float ang = (float)s * inv;
        float c = cosf(ang), sn = sinf(ang);
        float q1 = qr[d], q2 = qr[d + 32];
        q16[o16 + d]      = (_Float16)(q1 * c - q2 * sn);
        q16[o16 + d + 32] = (_Float16)(q2 * c + q1 * sn);
        float k1 = kr[d], k2 = kr[d + 32];
        k16[o16 + d]      = (_Float16)(k1 * c - k2 * sn);
        k16[o16 + d + 32] = (_Float16)(k2 * c + k1 * sn);
    }
    const size_t ov = ((size_t)(b * H + h) * HD) * S + s;
#pragma unroll 8
    for (int d = 0; d < HD; ++d) vT16[ov + (size_t)d * S] = (_Float16)vr[d];
}

// ---------------------------------------------------------------- flash attention
// 4 waves/block, one 16-row Q tile per wave; 32-key chunks; online softmax.
__global__ __launch_bounds__(128)
void attn_flash(const _Float16* __restrict__ Q, const _Float16* __restrict__ Kt,
                const _Float16* __restrict__ Vt, _Float16* __restrict__ O,
                int B, int H, int S) {
    const int HD = 64, D = 1024;
    int wv_ = threadIdx.x >> 5;
    int tile = blockIdx.x * 4 + wv_;
    int nqt = S / 16;
    int qt = tile % nqt;
    int h  = (tile / nqt) % H;
    int b  = tile / (nqt * H);
    int lane = threadIdx.x & 31;
    int g = lane >> 4, ln = lane & 15;
    int q0 = qt * 16;

    const _Float16* qbase = Q  + ((size_t)(b * H + h) * S + q0) * HD;
    const _Float16* kbase = Kt + ((size_t)(b * H + h) * S) * HD;   // [S][HD]
    const _Float16* vbase = Vt + ((size_t)(b * H + h) * HD) * S;   // [HD][S]

    v16h qa[2];
#pragma unroll
    for (int c = 0; c < 2; ++c) {
        const _Float16* qr = qbase + (size_t)ln * HD + 32 * c;
        v8h lo = *reinterpret_cast<const v8h*>(qr + 8 * g);
        v8h hi = *reinterpret_cast<const v8h*>(qr + 16 + 8 * g);
#pragma unroll
        for (int j = 0; j < 8; ++j) { qa[c][j] = lo[j]; qa[c][j + 8] = hi[j]; }
    }

    v8f o_acc[4] = {};
    float mrun[8], lrun[8];
#pragma unroll
    for (int r = 0; r < 8; ++r) { mrun[r] = -1e30f; lrun[r] = 0.f; }

    __shared__ __align__(16) _Float16 P[4][16 * 32];
    _Float16* Pw = P[wv_];

    for (int kk = 0; kk < S; kk += 32) {
        v8f s0 = {}, s1 = {};
        {
            const _Float16* kr0 = kbase + (size_t)(kk + ln) * HD;
            const _Float16* kr1 = kbase + (size_t)(kk + 16 + ln) * HD;
            v16h kb;
            auto ldk = [&](const _Float16* p) {
                v16h r;
                v8h lo = *reinterpret_cast<const v8h*>(p + 8 * g);
                v8h hi = *reinterpret_cast<const v8h*>(p + 16 + 8 * g);
#pragma unroll
                for (int j = 0; j < 8; ++j) { r[j] = lo[j]; r[j + 8] = hi[j]; }
                return r;
            };
            kb = ldk(kr0);
            s0 = wmma_f16(qa[0], kb, s0);
            kb = ldk(kr0 + 32);
            s0 = wmma_f16(qa[1], kb, s0);
            kb = ldk(kr1);
            s1 = wmma_f16(qa[0], kb, s1);
            kb = ldk(kr1 + 32);
            s1 = wmma_f16(qa[1], kb, s1);
        }
        const float scale = 0.125f;  // 1/sqrt(64)
        float alpha[8];
#pragma unroll
        for (int r = 0; r < 8; ++r) {
            float a = s0[r] * scale, c = s1[r] * scale;
            float mx = fmaxf(a, c);
            mx = fmaxf(mx, __shfl_xor(mx, 1, 32));
            mx = fmaxf(mx, __shfl_xor(mx, 2, 32));
            mx = fmaxf(mx, __shfl_xor(mx, 4, 32));
            mx = fmaxf(mx, __shfl_xor(mx, 8, 32));
            float mn = fmaxf(mrun[r], mx);
            float al = expf(mrun[r] - mn);
            float p0 = expf(a - mn), p1 = expf(c - mn);
            float rs = p0 + p1;
            rs += __shfl_xor(rs, 1, 32);
            rs += __shfl_xor(rs, 2, 32);
            rs += __shfl_xor(rs, 4, 32);
            rs += __shfl_xor(rs, 8, 32);
            lrun[r] = lrun[r] * al + rs;
            mrun[r] = mn;
            alpha[r] = al;
            Pw[(r + 8 * g) * 32 + ln]      = (_Float16)p0;
            Pw[(r + 8 * g) * 32 + 16 + ln] = (_Float16)p1;
        }
#pragma unroll
        for (int t = 0; t < 4; ++t)
#pragma unroll
            for (int r = 0; r < 8; ++r) o_acc[t][r] *= alpha[r];

        v16h pa;
        {
            v8h lo = *reinterpret_cast<const v8h*>(&Pw[ln * 32 + 8 * g]);
            v8h hi = *reinterpret_cast<const v8h*>(&Pw[ln * 32 + 16 + 8 * g]);
#pragma unroll
            for (int j = 0; j < 8; ++j) { pa[j] = lo[j]; pa[j + 8] = hi[j]; }
        }
#pragma unroll
        for (int t = 0; t < 4; ++t) {
            const _Float16* vr = vbase + (size_t)(16 * t + ln) * S + kk + 8 * g;
            v16h vb;
            v8h vlo = *reinterpret_cast<const v8h*>(vr);
            v8h vhi = *reinterpret_cast<const v8h*>(vr + 16);
#pragma unroll
            for (int j = 0; j < 8; ++j) { vb[j] = vlo[j]; vb[j + 8] = vhi[j]; }
            o_acc[t] = wmma_f16(pa, vb, o_acc[t]);
        }
    }

    _Float16* ob = O + ((size_t)b * S + q0) * D + h * HD;
#pragma unroll
    for (int r = 0; r < 8; ++r) {
        float inv = 1.f / lrun[r];
        int row = r + 8 * g;
#pragma unroll
        for (int t = 0; t < 4; ++t)
            ob[(size_t)row * D + 16 * t + ln] = (_Float16)(o_acc[t][r] * inv);
    }
}

// ---------------------------------------------------------------- host launcher
extern "C" void kernel_launch(void* const* d_in, const int* in_sizes, int n_in,
                              void* d_out, int out_size, void* d_ws, size_t ws_size,
                              hipStream_t stream) {
    const int B = 4, S = 2048, D = 1024, F = 4096, H = 16;
    const int M = B * S;

    const float* x    = (const float*)d_in[0];
    const float* ln1g = (const float*)d_in[2];
    const float* ln1b = (const float*)d_in[3];
    const float* ln2g = (const float*)d_in[4];
    const float* ln2b = (const float*)d_in[5];
    const float* wq   = (const float*)d_in[6];
    const float* wk   = (const float*)d_in[7];
    const float* wv   = (const float*)d_in[8];
    const float* wo   = (const float*)d_in[9];
    const float* w1   = (const float*)d_in[10];
    const float* b1   = (const float*)d_in[11];
    const float* w2   = (const float*)d_in[12];
    const float* b2   = (const float*)d_in[13];
    float* out = (float*)d_out;

    char* p = (char*)d_ws;
    auto alloc = [&](size_t bytes) -> char* {
        char* r = p; p += (bytes + 255) & ~(size_t)255; return r;
    };
    _Float16* wqkvT = (_Float16*)alloc((size_t)3 * D * D * 2);  // [3D][D]
    _Float16* woT   = (_Float16*)alloc((size_t)D * D * 2);      // [D][D]
    _Float16* w1T   = (_Float16*)alloc((size_t)F * D * 2);      // [F][D]
    _Float16* w2T   = (_Float16*)alloc((size_t)D * F * 2);      // [D][F]
    _Float16* h16   = (_Float16*)alloc((size_t)M * D * 2);      // reused for h2
    float*    x1    = (float*)   alloc((size_t)M * D * 4);
    _Float16* q16   = (_Float16*)alloc((size_t)M * D * 2);      // [B,H,S,HD]
    _Float16* k16   = (_Float16*)alloc((size_t)M * D * 2);
    _Float16* vT16  = (_Float16*)alloc((size_t)M * D * 2);      // [B,H,HD,S]
    _Float16* att16 = (_Float16*)alloc((size_t)M * D * 2);      // [B,S,D]
    float*    qkvf  = (float*)   alloc((size_t)M * 3 * D * 4);  // [M][3D]
    _Float16* ffn16 = (_Float16*)qkvf;   // alias (64MB over 96MB, dead by then)

    // 1. weights -> f16, transposed to [N][K]
    dim3 tDD(D / 32, D / 32);
    transpose_cast<<<tDD, 256, 0, stream>>>(wq, wqkvT,                     D, D);
    transpose_cast<<<tDD, 256, 0, stream>>>(wk, wqkvT + (size_t)D * D,     D, D);
    transpose_cast<<<tDD, 256, 0, stream>>>(wv, wqkvT + (size_t)2 * D * D, D, D);
    transpose_cast<<<tDD, 256, 0, stream>>>(wo, woT, D, D);
    transpose_cast<<<dim3(F / 32, D / 32), 256, 0, stream>>>(w1, w1T, D, F);
    transpose_cast<<<dim3(D / 32, F / 32), 256, 0, stream>>>(w2, w2T, F, D);

    // 2. LN1
    ln_cast<<<M, 256, 0, stream>>>(x, ln1g, ln1b, h16, D);

    // 3. fused QKV projection (WMMA, async-LDS pipeline): [M,3D]
    gemm_f16<<<dim3(3 * D / BN, M / BM), 256, 0, stream>>>(
        h16, wqkvT, nullptr, nullptr, qkvf, nullptr, M, 3 * D, D, 0);

    // 4. RoPE + relayout to f16
    rope_cast<<<(B * S * H + 255) / 256, 256, 0, stream>>>(qkvf, q16, k16, vT16, B, H, S);

    // 5. flash attention (WMMA)
    attn_flash<<<B * H * (S / 16) / 4, 128, 0, stream>>>(q16, k16, vT16, att16, B, H, S);

    // 6. out projection + residual -> x1
    gemm_f16<<<dim3(D / BN, M / BM), 256, 0, stream>>>(
        att16, woT, nullptr, x, x1, nullptr, M, D, D, 1);

    // 7. LN2
    ln_cast<<<M, 256, 0, stream>>>(x1, ln2g, ln2b, h16, D);

    // 8. FFN1: gelu(h2 @ w1 + b1) -> f16
    gemm_f16<<<dim3(F / BN, M / BM), 256, 0, stream>>>(
        h16, w1T, b1, nullptr, nullptr, ffn16, M, F, D, 2);

    // 9. FFN2: x1 + (ffn @ w2 + b2) -> out
    gemm_f16<<<dim3(D / BN, M / BM), 256, 0, stream>>>(
        ffn16, w2T, b2, x1, out, nullptr, M, D, F, 3);
}
